// SE3Attention_24610162606607
// MI455X (gfx1250) — matrix operand order repaired and checked by reading, not compiled, not run on hardware
//
#include <hip/hip_runtime.h>

#define HID 256
#define NTOK 4096
#define NHEAD 8
#define DH 32
#define KT 64  // keys staged per tile in the attention kernel

typedef __attribute__((ext_vector_type(16))) _Float16 v16h;
typedef __attribute__((ext_vector_type(8)))  _Float16 v8h;
typedef __attribute__((ext_vector_type(8)))  float    v8f;
typedef __attribute__((ext_vector_type(4)))  int      v4i;

// --------------------------------------------------------------------------
// gfx1250 async global->LDS path (guarded: falls back to sync ds stores)
// --------------------------------------------------------------------------
#if defined(__has_builtin)
#if __has_builtin(__builtin_amdgcn_global_load_async_to_lds_b128) && \
    __has_builtin(__builtin_amdgcn_s_wait_asynccnt)
#define USE_ASYNC_LDS 1
#endif
#endif
#ifndef USE_ASYNC_LDS
#define USE_ASYNC_LDS 0
#endif

__device__ __forceinline__ void async_copy_b128(void* lds, const void* g) {
#if USE_ASYNC_LDS
  __builtin_amdgcn_global_load_async_to_lds_b128(
      (__attribute__((address_space(1))) v4i*)g,
      (__attribute__((address_space(3))) v4i*)lds, 0, 0);
#else
  *(v8h*)lds = *(const v8h*)g;  // synchronous fallback (global load + ds store)
#endif
}

__device__ __forceinline__ void wait_async_prev() {
#if USE_ASYNC_LDS
  __builtin_amdgcn_s_wait_asynccnt(3);  // allow the 3 just-issued ops in flight
#endif
}
__device__ __forceinline__ void wait_async_all() {
#if USE_ASYNC_LDS
  __builtin_amdgcn_s_wait_asynccnt(0);
#endif
}

// ---------------------------------------------------------------------------
// WMMA wrapper: D = A(16x32 f16) * B(32x16 f16) + C(16x16 f32)
// ---------------------------------------------------------------------------
__device__ __forceinline__ v8f wmma_f16(v16h a, v16h b, v8f c) {
  return __builtin_amdgcn_wmma_f32_16x16x32_f16(
      false, a, false, b, (short)0, c, false, false);
}

// A-fragment (16x32, row-major source). Lane m (m<16): row m0+m, K = k0..k0+15
// contiguous; lane m+16: same row, K = k0+16..k0+31.
__device__ __forceinline__ v16h load_A(const _Float16* __restrict__ p, int ld,
                                       int m0, int k0, int lane) {
  const int half = lane >> 4;
  const int r    = lane & 15;
  const _Float16* row = p + (long)(m0 + r) * ld + k0 + half * 16;
  v16h a;
  ((v8h*)&a)[0] = *(const v8h*)(row);
  ((v8h*)&a)[1] = *(const v8h*)(row + 8);
  return a;
}

// B-fragment (32x16): B[k][n] = W[n0+n][k] with W row-major [Nout, K].
// Lane n: col n0+n, K = k0..k0+15 contiguous; lane n+16: K = k0+16..k0+31.
__device__ __forceinline__ v16h load_Bt(const _Float16* __restrict__ w, int ld,
                                        int k0, int n0, int lane) {
  const int half = lane >> 4;
  const int n    = lane & 15;
  return *(const v16h*)(w + (long)(n0 + n) * ld + k0 + half * 16);
}

// ---------------------------------------------------------------------------
// fp32 -> fp16 conversion
// ---------------------------------------------------------------------------
__global__ void f32_to_f16_kernel(const float* __restrict__ in,
                                  _Float16* __restrict__ out, int n) {
  int i = blockIdx.x * blockDim.x + threadIdx.x;
  if (i < n) out[i] = (_Float16)in[i];
}

// ---------------------------------------------------------------------------
// QKV projection: out = x @ W^T + b, stored per-head.
// transposeOut==0: [B,H,N,32] (Q,K); ==1: [B,H,32,N] (V, for P*V B-frags)
// ---------------------------------------------------------------------------
__global__ __launch_bounds__(256) void qkv_proj_kernel(
    const _Float16* __restrict__ xh, const _Float16* __restrict__ wh,
    const float* __restrict__ bias, _Float16* __restrict__ outh,
    int transposeOut) {
  const int lane = threadIdx.x & 31;
  const int wave = threadIdx.x >> 5;
  const int m0 = blockIdx.x * 64 + (wave >> 2) * 32;
  const int n0 = blockIdx.y * 128 + (wave & 3) * 32;

  v8f acc00 = {}, acc01 = {}, acc10 = {}, acc11 = {};
#pragma unroll
  for (int k0 = 0; k0 < HID; k0 += 32) {
    v16h a0 = load_A(xh, HID, m0, k0, lane);
    v16h a1 = load_A(xh, HID, m0 + 16, k0, lane);
    v16h b0 = load_Bt(wh, HID, k0, n0, lane);
    v16h b1 = load_Bt(wh, HID, k0, n0 + 16, lane);
    acc00 = wmma_f16(a0, b0, acc00);
    acc01 = wmma_f16(a0, b1, acc01);
    acc10 = wmma_f16(a1, b0, acc10);
    acc11 = wmma_f16(a1, b1, acc11);
  }

  const int hi = lane >> 4;
  const int ln = lane & 15;
#pragma unroll
  for (int im = 0; im < 2; ++im) {
#pragma unroll
    for (int in_ = 0; in_ < 2; ++in_) {
      const v8f& c = im ? (in_ ? acc11 : acc10) : (in_ ? acc01 : acc00);
      const int col = n0 + in_ * 16 + ln;
      const float bv = bias[col];
      const int h = col >> 5, d = col & 31;
#pragma unroll
      for (int r = 0; r < 8; ++r) {
        const int row = m0 + im * 16 + hi * 8 + r;
        const int b = row >> 12, t = row & (NTOK - 1);
        const _Float16 hv = (_Float16)(c[r] + bv);
        if (!transposeOut)
          outh[(((long)(b * NHEAD + h) * NTOK) + t) * DH + d] = hv;
        else
          outh[((long)(b * NHEAD + h) * DH + d) * NTOK + t] = hv;
      }
    }
  }
}

// ---------------------------------------------------------------------------
// Flash attention with SE(3) distance bias.
// Block = 8 waves sharing one (b,h); each wave owns 16 query rows.
// K/V^T/pos tiles (64 keys) are staged into LDS cooperatively and
// double-buffered via async global->LDS copies (ASYNCcnt).
// grid = (N/128, NHEAD, B).
// ---------------------------------------------------------------------------
__global__ __launch_bounds__(256) void se3_attn_kernel(
    const _Float16* __restrict__ qh,  // [B,H,N,32]
    const _Float16* __restrict__ kh,  // [B,H,N,32]
    const _Float16* __restrict__ vt,  // [B,H,32,N]
    const float* __restrict__ pos,    // [B,N,3]
    _Float16* __restrict__ outh) {    // [B,N,256]
  __shared__ _Float16 sK[2][KT * DH];    // 2 x 4 KB, [key][d]
  __shared__ _Float16 sV[2][DH * KT];    // 2 x 4 KB, [d][key]
  __shared__ float    sPos[2][KT * 3];   // 2 x 768 B
  __shared__ _Float16 sP[8][16 * KT];    // per-wave P staging, 8 x 2 KB

  const int tid  = threadIdx.x;
  const int lane = tid & 31;
  const int wave = tid >> 5;
  const int b = blockIdx.z, h = blockIdx.y;
  const int q0 = blockIdx.x * 128 + wave * 16;

  const _Float16* qhead = qh + (long)(b * NHEAD + h) * NTOK * DH;
  const _Float16* khead = kh + (long)(b * NHEAD + h) * NTOK * DH;
  const _Float16* vhead = vt + (long)(b * NHEAD + h) * DH * NTOK;
  const float* posb = pos + (long)b * NTOK * 3;

  // Stage one 64-key tile into buffer bb: every thread issues exactly 3
  // async ops so the per-wave ASYNCcnt is uniform (pos chunks duplicated).
  auto stage = [&](int j0, int bb) {
    // K tile: 64x32 f16 = 4 KB, 256 x 16 B chunks
    async_copy_b128(&sK[bb][tid * 8], khead + (long)j0 * DH + tid * 8);
    // V^T tile: 32 rows x (64 f16 = 128 B) = 4 KB
    const int d = tid >> 3, c = (tid & 7) * 8;
    async_copy_b128(&sV[bb][d * KT + c], vhead + (long)d * NTOK + j0 + c);
    // pos tile: 192 floats = 48 x 16 B (threads >=48 duplicate, same bytes)
    const int pc = (tid % 48) * 4;
    async_copy_b128(&sPos[bb][pc], posb + (long)j0 * 3 + pc);
  };

  // Q A-fragment: 16 rows x Dh=32, held in 8 VGPRs for the whole loop.
  const v16h aq = load_A(qhead, DH, q0, 0, lane);

  const int hi = lane >> 4;
  const int ln = lane & 15;

  float px[8], py[8], pz[8], qq[8], m[8], l[8];
#pragma unroll
  for (int r = 0; r < 8; ++r) {
    const int row = q0 + hi * 8 + r;
    px[r] = posb[row * 3 + 0];
    py[r] = posb[row * 3 + 1];
    pz[r] = posb[row * 3 + 2];
    qq[r] = px[r] * px[r] + py[r] * py[r] + pz[r] * pz[r];
    m[r] = -1e30f;
    l[r] = 0.0f;
  }

  v8f o0 = {}, o1 = {};
  const float inv_scale = 0.17677669529663687f;  // 1/sqrt(32)
  const v8f zero = {};
  _Float16* myP = &sP[wave][0];

  stage(0, 0);

  for (int j0 = 0; j0 < NTOK; j0 += KT) {
    const int bb = (j0 / KT) & 1;
    if (j0 + KT < NTOK) {
      stage(j0 + KT, bb ^ 1);
      wait_async_prev();  // previous stage's 3 ops retired
    } else {
      wait_async_all();
    }
    __syncthreads();  // buffer bb fully staged & visible block-wide

    const _Float16* kbuf = &sK[bb][0];
    const _Float16* vbuf = &sV[bb][0];
    const float*    pbuf = &sPos[bb][0];

    // --- scores: S = Q * K^T, four 16x16 tiles over 64 keys ---
    v8f s[4];
#pragma unroll
    for (int jt = 0; jt < 4; ++jt) {
      v16h bk = load_Bt(kbuf, DH, 0, jt * 16, lane);
      s[jt] = wmma_f16(aq, bk, zero);
    }

    // key-column positions for this lane (col = ln in every score tile)
    float kx[4], ky[4], kz[4], kk[4];
#pragma unroll
    for (int jt = 0; jt < 4; ++jt) {
      const int jc = jt * 16 + ln;
      kx[jt] = pbuf[jc * 3 + 0];
      ky[jt] = pbuf[jc * 3 + 1];
      kz[jt] = pbuf[jc * 3 + 2];
      kk[jt] = kx[jt] * kx[jt] + ky[jt] * ky[jt] + kz[jt] * kz[jt];
    }

    // --- bias + online softmax per C-layout row ---
#pragma unroll
    for (int r = 0; r < 8; ++r) {
      float t[4];
#pragma unroll
      for (int jt = 0; jt < 4; ++jt) {
        const float d2 = qq[r] + kk[jt] -
                         2.0f * (px[r] * kx[jt] + py[r] * ky[jt] + pz[r] * kz[jt]);
        t[jt] = s[jt][r] * inv_scale - 0.1f * sqrtf(fmaxf(d2, 0.0f));
      }
      float tm = fmaxf(fmaxf(t[0], t[1]), fmaxf(t[2], t[3]));
      tm = fmaxf(tm, __shfl_xor(tm, 1, 32));
      tm = fmaxf(tm, __shfl_xor(tm, 2, 32));
      tm = fmaxf(tm, __shfl_xor(tm, 4, 32));
      tm = fmaxf(tm, __shfl_xor(tm, 8, 32));
      const float mnew = fmaxf(m[r], tm);
      const float sc = __expf(m[r] - mnew);
      float e[4], es = 0.0f;
#pragma unroll
      for (int jt = 0; jt < 4; ++jt) {
        e[jt] = __expf(t[jt] - mnew);
        es += e[jt];
      }
      es += __shfl_xor(es, 1, 32);
      es += __shfl_xor(es, 2, 32);
      es += __shfl_xor(es, 4, 32);
      es += __shfl_xor(es, 8, 32);
      l[r] = l[r] * sc + es;
      m[r] = mnew;
      o0[r] *= sc;
      o1[r] *= sc;
      const int prow = hi * 8 + r;
#pragma unroll
      for (int jt = 0; jt < 4; ++jt)
        myP[prow * KT + jt * 16 + ln] = (_Float16)e[jt];
    }
    __builtin_amdgcn_wave_barrier();  // order wave-private ds stores vs loads

    // --- O += P * V: two K-steps of 32 keys x two d-tiles ---
    v16h ap0 = load_A(myP, KT, 0, 0, lane);
    v16h ap1 = load_A(myP, KT, 0, 32, lane);
    o0 = wmma_f16(ap0, load_Bt(vbuf, KT, 0, 0, lane), o0);
    o1 = wmma_f16(ap0, load_Bt(vbuf, KT, 0, 16, lane), o1);
    o0 = wmma_f16(ap1, load_Bt(vbuf, KT, 32, 0, lane), o0);
    o1 = wmma_f16(ap1, load_Bt(vbuf, KT, 32, 16, lane), o1);

    __syncthreads();  // all waves done with buffer bb before it is re-staged
  }

  // finalize: divide by softmax denom, write [B,N,H*32] f16
#pragma unroll
  for (int r = 0; r < 8; ++r) {
    const int row = q0 + hi * 8 + r;
    const float inv = 1.0f / l[r];
    _Float16* dst = outh + ((long)b * NTOK + row) * HID + h * DH;
    dst[ln]      = (_Float16)(o0[r] * inv);
    dst[ln + 16] = (_Float16)(o1[r] * inv);
  }
}

// ---------------------------------------------------------------------------
// Output projection: d_out = attn @ Wo^T + bo  (f32 output)
// ---------------------------------------------------------------------------
__global__ __launch_bounds__(256) void oproj_kernel(
    const _Float16* __restrict__ ah, const _Float16* __restrict__ wh,
    const float* __restrict__ bias, float* __restrict__ out) {
  const int lane = threadIdx.x & 31;
  const int wave = threadIdx.x >> 5;
  const int m0 = blockIdx.x * 64 + (wave >> 2) * 32;
  const int n0 = blockIdx.y * 128 + (wave & 3) * 32;

  v8f acc00 = {}, acc01 = {}, acc10 = {}, acc11 = {};
#pragma unroll
  for (int k0 = 0; k0 < HID; k0 += 32) {
    v16h a0 = load_A(ah, HID, m0, k0, lane);
    v16h a1 = load_A(ah, HID, m0 + 16, k0, lane);
    v16h b0 = load_Bt(wh, HID, k0, n0, lane);
    v16h b1 = load_Bt(wh, HID, k0, n0 + 16, lane);
    acc00 = wmma_f16(a0, b0, acc00);
    acc01 = wmma_f16(a0, b1, acc01);
    acc10 = wmma_f16(a1, b0, acc10);
    acc11 = wmma_f16(a1, b1, acc11);
  }

  const int hi = lane >> 4;
  const int ln = lane & 15;
#pragma unroll
  for (int im = 0; im < 2; ++im) {
#pragma unroll
    for (int in_ = 0; in_ < 2; ++in_) {
      const v8f& c = im ? (in_ ? acc11 : acc10) : (in_ ? acc01 : acc00);
      const int col = n0 + in_ * 16 + ln;
      const float bv = bias[col];
#pragma unroll
      for (int r = 0; r < 8; ++r) {
        const int row = m0 + im * 16 + hi * 8 + r;
        out[(long)row * HID + col] = c[r] + bv;
      }
    }
  }
}

// ---------------------------------------------------------------------------
// Host-side launch
// ---------------------------------------------------------------------------
extern "C" void kernel_launch(void* const* d_in, const int* in_sizes, int n_in,
                              void* d_out, int out_size, void* d_ws,
                              size_t ws_size, hipStream_t stream) {
  const float* x   = (const float*)d_in[0];
  const float* pos = (const float*)d_in[1];
  const float* Wq  = (const float*)d_in[2];
  const float* bq  = (const float*)d_in[3];
  const float* Wk  = (const float*)d_in[4];
  const float* bk  = (const float*)d_in[5];
  const float* Wv  = (const float*)d_in[6];
  const float* bv  = (const float*)d_in[7];
  const float* Wo  = (const float*)d_in[8];
  const float* bo  = (const float*)d_in[9];
  float* out = (float*)d_out;

  const int M  = 2 * NTOK;   // 8192 tokens
  const int NX = M * HID;    // 2,097,152
  const int NW = HID * HID;  // 65,536

  char* ws = (char*)d_ws;
  _Float16* xh  = (_Float16*)(ws);                // 4 MB
  _Float16* wqh = (_Float16*)(ws + (4u << 20));   // 128 KB each
  _Float16* wkh = wqh + NW;
  _Float16* wvh = wkh + NW;
  _Float16* woh = wvh + NW;
  _Float16* qh  = (_Float16*)(ws + (5u << 20));   // [B,H,N,32] 4 MB
  _Float16* kh  = (_Float16*)(ws + (9u << 20));   // [B,H,N,32] 4 MB
  _Float16* vt  = (_Float16*)(ws + (13u << 20));  // [B,H,32,N] 4 MB
  _Float16* ah  = (_Float16*)(ws + (17u << 20));  // [B,N,256]  4 MB

  f32_to_f16_kernel<<<(NX + 255) / 256, 256, 0, stream>>>(x, xh, NX);
  f32_to_f16_kernel<<<(NW + 255) / 256, 256, 0, stream>>>(Wq, wqh, NW);
  f32_to_f16_kernel<<<(NW + 255) / 256, 256, 0, stream>>>(Wk, wkh, NW);
  f32_to_f16_kernel<<<(NW + 255) / 256, 256, 0, stream>>>(Wv, wvh, NW);
  f32_to_f16_kernel<<<(NW + 255) / 256, 256, 0, stream>>>(Wo, woh, NW);

  dim3 gProj(M / 64, HID / 128);
  qkv_proj_kernel<<<gProj, 256, 0, stream>>>(xh, wqh, bq, qh, 0);
  qkv_proj_kernel<<<gProj, 256, 0, stream>>>(xh, wkh, bk, kh, 0);
  qkv_proj_kernel<<<gProj, 256, 0, stream>>>(xh, wvh, bv, vt, 1);

  dim3 gAttn(NTOK / 128, NHEAD, 2);
  se3_attn_kernel<<<gAttn, 256, 0, stream>>>(qh, kh, vt, pos, ah);

  oproj_kernel<<<gProj, 256, 0, stream>>>(ah, woh, bo, out);
}
